// TransformerBlock_21646635171948
// MI455X (gfx1250) — matrix operand order repaired and checked by reading, not compiled, not run on hardware
//
#include <hip/hip_runtime.h>
#include <math.h>

#define D_MODEL 1024
#define NHEAD   16
#define DHEAD   64
#define BATCH   4
#define NSEQL   1024
#define LQ      1025      // NSEQ + 1 struct token
#define LPAD    1056      // padded key length (multiple of 32)
#define NQT     65        // ceil(1025/16) query tiles of 16

typedef __attribute__((ext_vector_type(16))) _Float16 v16h;
typedef __attribute__((ext_vector_type(8)))  _Float16 v8h;
typedef __attribute__((ext_vector_type(8)))  float    v8f;

#ifndef __has_builtin
#define __has_builtin(x) 0
#endif
#if __has_builtin(__builtin_amdgcn_global_load_async_to_lds_b128)
#define HAVE_ASYNC 1
#else
#define HAVE_ASYNC 0
#endif

#if HAVE_ASYNC
// exact parameter pointee type per hipcc diagnostic: int __vector(16 bytes)
typedef int a_v4i __attribute__((vector_size(16)));
typedef __attribute__((address_space(1))) a_v4i gas_v4i;
typedef __attribute__((address_space(3))) a_v4i las_v4i;
#if __has_builtin(__builtin_amdgcn_s_wait_asynccnt)
#define WAIT_ASYNC(n) __builtin_amdgcn_s_wait_asynccnt(n)
#else
#define WAIT_ASYNC(n) asm volatile("s_wait_asynccnt %0" ::"n"(n) : "memory")
#endif
#endif

// ---------------------------------------------------------------- WMMA helpers
__device__ inline v8f wmma16(v16h a, v16h b, v8f c) {
  return __builtin_amdgcn_wmma_f32_16x16x32_f16(false, a, false, b, (short)0, c,
                                                false, false);
}

// A fragment 16x32 f16 from row-major src (lda in halves).
// lane l: row m = l&15; halves 0..15 -> K = hi*16 + {0..15}
__device__ inline v16h load_afrag(const _Float16* base, int lda, int lane) {
  int m = lane & 15, hi = lane >> 4;
  const _Float16* p = base + (size_t)m * lda + hi * 16;
  union { v16h v; v8h h[2]; } u;
  u.h[0] = *reinterpret_cast<const v8h*>(p);
  u.h[1] = *reinterpret_cast<const v8h*>(p + 8);
  return u.v;
}

// 16 contiguous halves per lane (packed-B or row-fragment loads)
__device__ inline v16h load_h16(const _Float16* p) {
  union { v16h v; v8h h[2]; } u;
  u.h[0] = *reinterpret_cast<const v8h*>(p);
  u.h[1] = *reinterpret_cast<const v8h*>(p + 8);
  return u.v;
}

// B fragment 32x16 where lane n = N row in memory, contiguous K run per lane:
// lane l: n = l&15, halves h -> K = hi*16 + h  (rows at stride ldb)
__device__ inline v16h load_bfrag(const _Float16* base, int ldb, int lane) {
  int n = lane & 15, hi = lane >> 4;
  return load_h16(base + (size_t)n * ldb + hi * 16);
}

#if HAVE_ASYNC
// async-DMA one 32x32 f16 A tile (2KB) into LDS: 4 x b128 per lane (row = lane)
__device__ inline void stage_a_async(const _Float16* gsrc, int ldk, _Float16* lds,
                                     int lane) {
  const _Float16* g = gsrc + (size_t)lane * ldk;
  _Float16* l = lds + lane * 32;
  #pragma unroll
  for (int i = 0; i < 4; ++i) {
    __builtin_amdgcn_global_load_async_to_lds_b128(
        (gas_v4i*)(g + i * 8), (las_v4i*)(l + i * 8), 0, 0);
  }
}
#endif

__device__ inline float gelu_f(float x) {
  return 0.5f * x * (1.0f + erff(x * 0.70710678118654752f));
}
__device__ inline float silu_f(float x) { return x / (1.0f + __expf(-x)); }

__device__ inline float2 block_sum2(float s, float s2, float* red) {
  #pragma unroll
  for (int off = 16; off >= 1; off >>= 1) {
    s  += __shfl_xor(s,  off, 32);
    s2 += __shfl_xor(s2, off, 32);
  }
  int w = threadIdx.x >> 5;
  if ((threadIdx.x & 31) == 0) { red[w * 2] = s; red[w * 2 + 1] = s2; }
  __syncthreads();
  float ts = 0.f, t2 = 0.f;
  #pragma unroll
  for (int i = 0; i < 8; ++i) { ts += red[i * 2]; t2 += red[i * 2 + 1]; }
  __syncthreads();
  return make_float2(ts, t2);
}

// ---------------------------------------------------------------- weight pack
// P[((kt*NT+nt)*32 + lane)*16 + h] = W[kt*32 + (lane>>4)*16 + h][nt*16 + (lane&15)]
__global__ void pack_w(const float* __restrict__ W, _Float16* __restrict__ P,
                       int K, int N) {
  int id = blockIdx.x * 256 + threadIdx.x;
  if (id >= K * N) return;
  int h = id & 15;
  int lane = (id >> 4) & 31;
  int rest = id >> 9;
  int NT = N >> 4;
  int nt = rest % NT, kt = rest / NT;
  int kk = kt * 32 + (lane >> 4) * 16 + h;
  int nn = nt * 16 + (lane & 15);
  P[id] = (_Float16)W[(size_t)kk * N + nn];
}

// ---------------------------------------------------------------- layernorm
__global__ __launch_bounds__(256) void ln_rows_f16(const float* __restrict__ X,
    const float* __restrict__ g, const float* __restrict__ be,
    _Float16* __restrict__ Y) {
  __shared__ float red[16];
  int row = blockIdx.x;
  const float* x = X + (size_t)row * D_MODEL;
  float s = 0.f, s2 = 0.f;
  for (int i = threadIdx.x; i < D_MODEL; i += 256) { float v = x[i]; s += v; s2 += v * v; }
  float2 t = block_sum2(s, s2, red);
  float mu = t.x * (1.0f / D_MODEL);
  float var = t.y * (1.0f / D_MODEL) - mu * mu;
  float rs = rsqrtf(var + 1e-5f);
  _Float16* y = Y + (size_t)row * D_MODEL;
  for (int i = threadIdx.x; i < D_MODEL; i += 256) {
    float v = (x[i] - mu) * rs * g[i];
    if (be) v += be[i];
    y[i] = (_Float16)v;
  }
}

__global__ __launch_bounds__(256) void ln_rows_f32(const float* __restrict__ X,
    const float* __restrict__ g, const float* __restrict__ be,
    float* __restrict__ Y) {
  __shared__ float red[16];
  int row = blockIdx.x;
  const float* x = X + (size_t)row * D_MODEL;
  float s = 0.f, s2 = 0.f;
  for (int i = threadIdx.x; i < D_MODEL; i += 256) { float v = x[i]; s += v; s2 += v * v; }
  float2 t = block_sum2(s, s2, red);
  float mu = t.x * (1.0f / D_MODEL);
  float var = t.y * (1.0f / D_MODEL) - mu * mu;
  float rs = rsqrtf(var + 1e-5f);
  float* y = Y + (size_t)row * D_MODEL;
  for (int i = threadIdx.x; i < D_MODEL; i += 256) {
    float v = (x[i] - mu) * rs * g[i];
    if (be) v += be[i];
    y[i] = v;
  }
}

// ---------------------------------------------------------------- WMMA GEMM
// C[M][N] = A_f16[M][K] * B_packed ; each wave -> 32x64 output:
// 2 A-fragments x 4 B-fragments -> 8 WMMAs per 32-deep K step.
// A tiles double-buffered in LDS via GLOBAL_LOAD_ASYNC_TO_LDS_B128 (ASYNCcnt),
// B fragments direct b128 register loads from the packed layout.
// mode 0: f32 store   1: f32 store + residual   2: f16 store w/ exact GELU
__global__ __launch_bounds__(256) void gemm_wmma(const _Float16* __restrict__ A,
    const _Float16* __restrict__ Bp, int M, int N, int K,
    void* __restrict__ Cout, const float* __restrict__ resid, int mode) {
#if HAVE_ASYNC
  __shared__ __align__(32) _Float16 Asmem[8][2][32 * 32];
#endif
  int gtid = blockIdx.x * 256 + threadIdx.x;
  int wid = gtid >> 5;
  int lane = threadIdx.x & 31;
  int NP = N >> 6;                 // 64-wide n tiles
  int totalW = (M >> 5) * NP;
  if (wid >= totalW) return;
  int mt = wid / NP, np = wid % NP;
  int NT = N >> 4;
  int KT = K >> 5;
  v8f acc[2][4] = {};
  const _Float16* arow = A + (size_t)mt * 32 * K;
  const _Float16* bbase = Bp + (size_t)np * 4 * 512 + (size_t)lane * 16;
  size_t bstep = (size_t)NT * 512;

#if HAVE_ASYNC
  int wvb = threadIdx.x >> 5;
  _Float16* sm[2] = { &Asmem[wvb][0][0], &Asmem[wvb][1][0] };
  stage_a_async(arow, K, sm[0], lane);
  for (int kt = 0; kt < KT; ++kt) {
    _Float16* cur = sm[kt & 1];
    if (kt + 1 < KT) {
      stage_a_async(arow + (kt + 1) * 32, K, sm[(kt + 1) & 1], lane);
      WAIT_ASYNC(4);               // current tile's 4 async ops retired
    } else {
      WAIT_ASYNC(0);
    }
    v16h a0 = load_afrag(cur, 32, lane);
    v16h a1 = load_afrag(cur + 16 * 32, 32, lane);
    const _Float16* bk = bbase + (size_t)kt * bstep;
    #pragma unroll
    for (int j = 0; j < 4; ++j) {
      v16h b = load_h16(bk + j * 512);
      acc[0][j] = wmma16(a0, b, acc[0][j]);
      acc[1][j] = wmma16(a1, b, acc[1][j]);
    }
  }
#else
  for (int kt = 0; kt < KT; ++kt) {
    v16h a0 = load_afrag(arow + kt * 32, K, lane);
    v16h a1 = load_afrag(arow + (size_t)16 * K + kt * 32, K, lane);
    const _Float16* bk = bbase + (size_t)kt * bstep;
    #pragma unroll
    for (int j = 0; j < 4; ++j) {
      v16h b = load_h16(bk + j * 512);
      acc[0][j] = wmma16(a0, b, acc[0][j]);
      acc[1][j] = wmma16(a1, b, acc[1][j]);
    }
  }
#endif

  int col = lane & 15, hi = lane >> 4;
  #pragma unroll
  for (int m = 0; m < 2; ++m) {
    #pragma unroll
    for (int j = 0; j < 4; ++j) {
      #pragma unroll
      for (int v = 0; v < 8; ++v) {
        int row = mt * 32 + m * 16 + v + hi * 8;
        size_t o = (size_t)row * N + np * 64 + j * 16 + col;
        float x = acc[m][j][v];
        if (mode == 1) x += resid[o];
        if (mode == 2) ((_Float16*)Cout)[o] = (_Float16)gelu_f(x);
        else           ((float*)Cout)[o] = x;
      }
    }
  }
}

// ---------------------------------------------------------------- QKV post
// per (b,t) row: q/k layernorm (full D), RoPE per head, write f16 Q/K rows and Vt
__global__ __launch_bounds__(256) void qkv_post(const float* __restrict__ QKV,
    const float* __restrict__ qg, const float* __restrict__ kg,
    _Float16* __restrict__ Q, _Float16* __restrict__ Kd,
    _Float16* __restrict__ Vt) {
  __shared__ float red[16];
  __shared__ float buf[D_MODEL];
  int row = blockIdx.x;
  int b = row >> 10, tpos = row & 1023;
  const float* src = QKV + (size_t)row * 3072;
  for (int part = 0; part < 2; ++part) {
    const float* x = src + part * D_MODEL;
    const float* g = part ? kg : qg;
    float s = 0.f, s2 = 0.f;
    for (int i = threadIdx.x; i < D_MODEL; i += 256) { float v = x[i]; s += v; s2 += v * v; }
    float2 t = block_sum2(s, s2, red);
    float mu = t.x * (1.0f / D_MODEL);
    float var = t.y * (1.0f / D_MODEL) - mu * mu;
    float rs = rsqrtf(var + 1e-5f);
    for (int i = threadIdx.x; i < D_MODEL; i += 256) buf[i] = (x[i] - mu) * rs * g[i];
    __syncthreads();
    _Float16* dst = part ? Kd : Q;
    for (int i = threadIdx.x; i < D_MODEL; i += 256) {
      int hh = i >> 6, r = i & 63, fi = r & 31, hf = r >> 5;
      float invf = __powf(10000.0f, -(float)fi * (1.0f / 32.0f));
      float ang = (float)tpos * invf;
      float sn, c;
      __sincosf(ang, &sn, &c);
      float partner = buf[(i & ~63) + (hf ? r - 32 : r + 32)];
      float val = hf ? buf[i] * c + partner * sn : buf[i] * c - partner * sn;
      dst[((size_t)(b * NHEAD + hh) * LPAD + tpos) * DHEAD + r] = (_Float16)val;
    }
    __syncthreads();
  }
  for (int i = threadIdx.x; i < D_MODEL; i += 256) {
    float v = src[2048 + i];
    int hh = i >> 6, r = i & 63;
    size_t bh = (size_t)(b * NHEAD + hh);
    Vt[(bh * DHEAD + r) * LPAD + tpos] = (_Float16)v;
  }
}

__global__ void zeropad_kv(_Float16* __restrict__ Q, _Float16* __restrict__ Kd,
                           _Float16* __restrict__ Vt) {
  int id = blockIdx.x * 256 + threadIdx.x;
  const int NPAD = LPAD - LQ; // 31
  int total = BATCH * NHEAD * NPAD * DHEAD;
  if (id >= total) return;
  int r = id & 63;
  int rest = id >> 6;
  int t = LQ + (rest % NPAD);
  int bh = rest / NPAD;
  size_t o = ((size_t)bh * LPAD + t) * DHEAD + r;
  Q[o] = (_Float16)0.f;
  Kd[o] = (_Float16)0.f;
  Vt[((size_t)bh * DHEAD + r) * LPAD + t] = (_Float16)0.f;
}

// ---------------------------------------------------------------- struct path
// timestep embedding for T_input=0: sin half = 0, cos half = 1
__global__ void s_hT(const float* __restrict__ Tw1, const float* __restrict__ Tb1,
                     float* __restrict__ hT) {
  int d = blockIdx.x * 256 + threadIdx.x;
  if (d >= D_MODEL) return;
  float a = Tb1[d];
  for (int j = 0; j < 128; ++j) a += Tw1[(size_t)(128 + j) * D_MODEL + d];
  hT[d] = silu_f(a);
}

__global__ void s_htemp(const float* __restrict__ temp, const float* __restrict__ tw1,
                        const float* __restrict__ tb1, float* __restrict__ ht) {
  int id = blockIdx.x * 256 + threadIdx.x;
  if (id >= BATCH * D_MODEL) return;
  int b = id >> 10, d = id & 1023;
  ht[id] = silu_f(temp[b] * tw1[d] + tb1[d]);
}

__global__ void s_Te(const float* __restrict__ hT, const float* __restrict__ Tw2,
                     const float* __restrict__ Tb2, float* __restrict__ Te) {
  int d = blockIdx.x * 256 + threadIdx.x;
  if (d >= D_MODEL) return;
  float a = Tb2[d];
  for (int k = 0; k < D_MODEL; ++k) a += hT[k] * Tw2[(size_t)k * D_MODEL + d];
  Te[d] = a;
}

__global__ void s_tempE(const float* __restrict__ ht, const float* __restrict__ tw2,
                        const float* __restrict__ tb2, float* __restrict__ tE) {
  int id = blockIdx.x * 256 + threadIdx.x;
  if (id >= BATCH * D_MODEL) return;
  int b = id >> 10, d = id & 1023;
  float a = tb2[d];
  for (int k = 0; k < D_MODEL; ++k) a += ht[b * D_MODEL + k] * tw2[(size_t)k * D_MODEL + d];
  tE[id] = a;
}

__global__ __launch_bounds__(256) void s_xmod(const float* __restrict__ xstruc,
    const float* __restrict__ g, const float* __restrict__ be,
    const float* __restrict__ tE, const float* __restrict__ Te,
    float* __restrict__ xmod) {
  __shared__ float red[16];
  int b = blockIdx.x;
  const float* x = xstruc + (size_t)b * D_MODEL;
  float s = 0.f, s2 = 0.f;
  for (int i = threadIdx.x; i < D_MODEL; i += 256) { float v = x[i]; s += v; s2 += v * v; }
  float2 t = block_sum2(s, s2, red);
  float mu = t.x * (1.0f / D_MODEL);
  float var = t.y * (1.0f / D_MODEL) - mu * mu;
  float rs = rsqrtf(var + 1e-5f);
  for (int i = threadIdx.x; i < D_MODEL; i += 256)
    xmod[b * D_MODEL + i] = (x[i] - mu) * rs * g[i] + be[i] + tE[b * D_MODEL + i] + Te[i];
}

__global__ void s_qkvt(const float* __restrict__ xmod, const float* __restrict__ W,
                       float* __restrict__ qkvt) {
  int id = blockIdx.x * 256 + threadIdx.x;
  if (id >= BATCH * 3072) return;
  int b = id / 3072, n = id % 3072;
  float a = 0.f;
  for (int k = 0; k < D_MODEL; ++k) a += xmod[b * D_MODEL + k] * W[(size_t)k * 3072 + n];
  qkvt[id] = a;
}

// q/k LN, RoPE@pos0 = identity, write struct token row (t = 1024)
__global__ __launch_bounds__(256) void struct_finish(const float* __restrict__ qkvt,
    const float* __restrict__ qg, const float* __restrict__ kg,
    _Float16* __restrict__ Q, _Float16* __restrict__ Kd, _Float16* __restrict__ Vt) {
  __shared__ float red[16];
  int b = blockIdx.x;
  const float* src = qkvt + (size_t)b * 3072;
  for (int part = 0; part < 3; ++part) {
    const float* x = src + part * D_MODEL;
    float mu = 0.f, rs = 1.f;
    if (part < 2) {
      float s = 0.f, s2 = 0.f;
      for (int i = threadIdx.x; i < D_MODEL; i += 256) { float v = x[i]; s += v; s2 += v * v; }
      float2 t = block_sum2(s, s2, red);
      mu = t.x * (1.0f / D_MODEL);
      float var = t.y * (1.0f / D_MODEL) - mu * mu;
      rs = rsqrtf(var + 1e-5f);
    }
    for (int i = threadIdx.x; i < D_MODEL; i += 256) {
      float val = (part == 0) ? (x[i] - mu) * rs * qg[i]
                : (part == 1) ? (x[i] - mu) * rs * kg[i]
                              : x[i];
      int hh = i >> 6, r = i & 63;
      size_t bh = (size_t)(b * NHEAD + hh);
      if (part == 0) Q[(bh * LPAD + NSEQL) * DHEAD + r] = (_Float16)val;
      else if (part == 1) Kd[(bh * LPAD + NSEQL) * DHEAD + r] = (_Float16)val;
      else Vt[(bh * DHEAD + r) * LPAD + NSEQL] = (_Float16)val;
    }
    __syncthreads();
  }
}

// ---------------------------------------------------------------- attention
// one wave per (b, h, 16-query tile); flash-style online softmax
__global__ __launch_bounds__(32) void attn_kernel(const _Float16* __restrict__ Q,
    const _Float16* __restrict__ Kd, const _Float16* __restrict__ Vt,
    _Float16* __restrict__ ctx_seq, _Float16* __restrict__ ctx_struc) {
  __shared__ __align__(32) _Float16 Pl[16 * 32];
  int lane = threadIdx.x;
  int qt = blockIdx.x % NQT;
  int bh = blockIdx.x / NQT;
  int h = bh & 15, b = bh >> 4;
  int q0 = qt * 16;
  const _Float16* Qb = Q + (size_t)bh * LPAD * DHEAD;
  const _Float16* Kb = Kd + (size_t)bh * LPAD * DHEAD;
  const _Float16* Vb = Vt + (size_t)bh * DHEAD * LPAD;

  v16h qa0 = load_afrag(Qb + (size_t)q0 * DHEAD, DHEAD, lane);
  v16h qa1 = load_afrag(Qb + (size_t)q0 * DHEAD + 32, DHEAD, lane);
  v8f o0 = {}, o1 = {}, o2 = {}, o3 = {};
  float mrow[8], lrow[8];
  #pragma unroll
  for (int v = 0; v < 8; ++v) { mrow[v] = -3.0e38f; lrow[v] = 0.f; }
  int col = lane & 15, hi = lane >> 4;
  int kend = (h == 0) ? LQ : (q0 + 16 < LQ ? q0 + 16 : LQ);

  for (int j0 = 0; j0 < kend; j0 += 32) {
    v8f s0 = {}, s1 = {};
    v16h kb;
    kb = load_bfrag(Kb + (size_t)j0 * DHEAD, DHEAD, lane);             s0 = wmma16(qa0, kb, s0);
    kb = load_bfrag(Kb + (size_t)j0 * DHEAD + 32, DHEAD, lane);        s0 = wmma16(qa1, kb, s0);
    kb = load_bfrag(Kb + (size_t)(j0 + 16) * DHEAD, DHEAD, lane);      s1 = wmma16(qa0, kb, s1);
    kb = load_bfrag(Kb + (size_t)(j0 + 16) * DHEAD + 32, DHEAD, lane); s1 = wmma16(qa1, kb, s1);

    #pragma unroll
    for (int v = 0; v < 8; ++v) {
      int qr = q0 + v + hi * 8;
      int kc0 = j0 + col, kc1 = j0 + 16 + col;
      float x0 = s0[v] * 0.125f, x1 = s1[v] * 0.125f;
      bool ok0 = (kc0 < LQ) && (qr < LQ) &&
                 ((h == 0 && qr < NSEQL && kc0 < NSEQL) || (kc0 <= qr));
      bool ok1 = (kc1 < LQ) && (qr < LQ) &&
                 ((h == 0 && qr < NSEQL && kc1 < NSEQL) || (kc1 <= qr));
      x0 = ok0 ? x0 : -1.0e9f;
      x1 = ok1 ? x1 : -1.0e9f;
      float mx = fmaxf(x0, x1);
      #pragma unroll
      for (int off = 8; off >= 1; off >>= 1) mx = fmaxf(mx, __shfl_xor(mx, off, 16));
      float mnew = fmaxf(mrow[v], mx);
      float scl = __expf(mrow[v] - mnew);
      float p0 = __expf(x0 - mnew), p1 = __expf(x1 - mnew);
      float ls = p0 + p1;
      #pragma unroll
      for (int off = 8; off >= 1; off >>= 1) ls += __shfl_xor(ls, off, 16);
      lrow[v] = lrow[v] * scl + ls;
      mrow[v] = mnew;
      o0[v] *= scl; o1[v] *= scl; o2[v] *= scl; o3[v] *= scl;
      int r = v + hi * 8;
      Pl[r * 32 + col] = (_Float16)p0;
      Pl[r * 32 + 16 + col] = (_Float16)p1;
    }
    // same-wave LDS ops are in order; compiler inserts DScnt waits for the RAW
    v16h pa = load_afrag((const _Float16*)Pl, 32, lane);
    v16h vb;
    vb = load_bfrag(Vb + (size_t)0 * LPAD + j0, LPAD, lane);  o0 = wmma16(pa, vb, o0);
    vb = load_bfrag(Vb + (size_t)16 * LPAD + j0, LPAD, lane); o1 = wmma16(pa, vb, o1);
    vb = load_bfrag(Vb + (size_t)32 * LPAD + j0, LPAD, lane); o2 = wmma16(pa, vb, o2);
    vb = load_bfrag(Vb + (size_t)48 * LPAD + j0, LPAD, lane); o3 = wmma16(pa, vb, o3);
  }

  #pragma unroll
  for (int v = 0; v < 8; ++v) {
    int qr = q0 + v + hi * 8;
    if (qr >= LQ) continue;
    float inv = 1.0f / lrow[v];
    _Float16* dst = (qr < NSEQL)
        ? ctx_seq + ((size_t)(b * NSEQL + qr) * D_MODEL + h * DHEAD)
        : ctx_struc + ((size_t)b * D_MODEL + h * DHEAD);
    dst[0 + col]  = (_Float16)(o0[v] * inv);
    dst[16 + col] = (_Float16)(o1[v] * inv);
    dst[32 + col] = (_Float16)(o2[v] * inv);
    dst[48 + col] = (_Float16)(o3[v] * inv);
  }
}

// ---------------------------------------------------------------- struct tail
__global__ void sdot_outstruc(const _Float16* __restrict__ ctxs,
    const float* __restrict__ W, const float* __restrict__ xstruc,
    float* __restrict__ xs2) {
  int id = blockIdx.x * 256 + threadIdx.x;
  if (id >= BATCH * D_MODEL) return;
  int b = id >> 10, d = id & 1023;
  float a = 0.f;
  for (int k = 0; k < D_MODEL; ++k)
    a += (float)ctxs[b * D_MODEL + k] * W[(size_t)k * D_MODEL + d];
  xs2[id] = a + xstruc[id];
}

__global__ void sffn1(const float* __restrict__ sln, const float* __restrict__ W1,
                      float* __restrict__ hbuf) {
  int id = blockIdx.x * 256 + threadIdx.x;
  if (id >= BATCH * 4096) return;
  int b = id >> 12, j = id & 4095;
  float a = 0.f;
  for (int k = 0; k < D_MODEL; ++k) a += sln[b * D_MODEL + k] * W1[(size_t)k * 4096 + j];
  hbuf[id] = gelu_f(a);
}

__global__ void sffn2(const float* __restrict__ hbuf, const float* __restrict__ W2,
                      const float* __restrict__ xs2, float* __restrict__ outp) {
  int id = blockIdx.x * 256 + threadIdx.x;
  if (id >= BATCH * D_MODEL) return;
  int b = id >> 10, d = id & 1023;
  float a = 0.f;
  for (int k = 0; k < 4096; ++k) a += hbuf[b * 4096 + k] * W2[(size_t)k * D_MODEL + d];
  outp[id] = xs2[id] + a;
}

// ---------------------------------------------------------------- launcher
extern "C" void kernel_launch(void* const* d_in, const int* in_sizes, int n_in,
                              void* d_out, int out_size, void* d_ws, size_t ws_size,
                              hipStream_t stream) {
  (void)in_sizes; (void)n_in; (void)out_size; (void)ws_size;
  const float* x_seq        = (const float*)d_in[0];
  const float* x_struc      = (const float*)d_in[1];
  const float* temp         = (const float*)d_in[2];
  const float* ln_qs_w      = (const float*)d_in[4];
  const float* ln_qs_b      = (const float*)d_in[5];
  const float* W_qkv_seq    = (const float*)d_in[6];
  const float* ln_qt_w      = (const float*)d_in[7];
  const float* ln_qt_b      = (const float*)d_in[8];
  const float* W_qkv_struc  = (const float*)d_in[9];
  const float* q_ln_seq_w   = (const float*)d_in[10];
  const float* k_ln_seq_w   = (const float*)d_in[11];
  const float* q_ln_struc_w = (const float*)d_in[12];
  const float* k_ln_struc_w = (const float*)d_in[13];
  const float* W_out_seq    = (const float*)d_in[14];
  const float* W_out_struc  = (const float*)d_in[15];
  const float* temp_w1      = (const float*)d_in[16];
  const float* temp_b1      = (const float*)d_in[17];
  const float* temp_w2      = (const float*)d_in[18];
  const float* temp_b2      = (const float*)d_in[19];
  const float* T_w1         = (const float*)d_in[20];
  const float* T_b1         = (const float*)d_in[21];
  const float* T_w2         = (const float*)d_in[22];
  const float* T_b2         = (const float*)d_in[23];
  const float* ffn_seq_ln_w = (const float*)d_in[24];
  const float* ffn_seq_ln_b = (const float*)d_in[25];
  const float* ffn_seq_w1   = (const float*)d_in[26];
  const float* ffn_seq_w2   = (const float*)d_in[27];
  const float* ffn_st_ln_w  = (const float*)d_in[28];
  const float* ffn_st_ln_b  = (const float*)d_in[29];
  const float* ffn_st_w1    = (const float*)d_in[30];
  const float* ffn_st_w2    = (const float*)d_in[31];

  char* ws = (char*)d_ws;
  size_t off = 0;
  auto alloc = [&](size_t bytes) -> size_t {
    size_t o = off;
    off += (bytes + 255) & ~(size_t)255;
    return o;
  };
  _Float16* wqkv_p = (_Float16*)(ws + alloc(1024ull * 3072 * 2));
  _Float16* wout_p = (_Float16*)(ws + alloc(1024ull * 1024 * 2));
  _Float16* w1_p   = (_Float16*)(ws + alloc(1024ull * 4096 * 2));
  _Float16* w2_p   = (_Float16*)(ws + alloc(4096ull * 1024 * 2));
  _Float16* xln    = (_Float16*)(ws + alloc(4096ull * 1024 * 2));
  float*    qkvbuf = (float*)(ws + alloc(4096ull * 3072 * 4));
  _Float16* hidden = (_Float16*)qkvbuf;  // reuse: qkv dead after qkv_post
  _Float16* Qf  = (_Float16*)(ws + alloc((size_t)BATCH * NHEAD * LPAD * DHEAD * 2));
  _Float16* Kf  = (_Float16*)(ws + alloc((size_t)BATCH * NHEAD * LPAD * DHEAD * 2));
  _Float16* Vtf = (_Float16*)(ws + alloc((size_t)BATCH * NHEAD * DHEAD * LPAD * 2));
  _Float16* ctxs = (_Float16*)(ws + alloc(4096ull * 1024 * 2));
  _Float16* ctxt = (_Float16*)(ws + alloc(4ull * 1024 * 2));
  float* hT    = (float*)(ws + alloc(1024ull * 4));
  float* htmp  = (float*)(ws + alloc(4096ull * 4));
  float* TeB   = (float*)(ws + alloc(1024ull * 4));
  float* tmpE  = (float*)(ws + alloc(4096ull * 4));
  float* xmod  = (float*)(ws + alloc(4096ull * 4));
  float* qkvt  = (float*)(ws + alloc(4ull * 3072 * 4));
  float* xs2   = (float*)(ws + alloc(4096ull * 4));
  float* slnb  = (float*)(ws + alloc(4096ull * 4));
  float* shid  = (float*)(ws + alloc(4ull * 4096 * 4));

  float* out_seq = (float*)d_out;
  float* out_st  = (float*)d_out + (size_t)BATCH * NSEQL * D_MODEL;

  // weight conversion / packing
  pack_w<<<(1024 * 3072 + 255) / 256, 256, 0, stream>>>(W_qkv_seq, wqkv_p, 1024, 3072);
  pack_w<<<(1024 * 1024 + 255) / 256, 256, 0, stream>>>(W_out_seq, wout_p, 1024, 1024);
  pack_w<<<(1024 * 4096 + 255) / 256, 256, 0, stream>>>(ffn_seq_w1, w1_p, 1024, 4096);
  pack_w<<<(4096 * 1024 + 255) / 256, 256, 0, stream>>>(ffn_seq_w2, w2_p, 4096, 1024);

  // seq: LN -> QKV GEMM -> q/k LN + RoPE -> f16 Q/K/Vt
  ln_rows_f16<<<4096, 256, 0, stream>>>(x_seq, ln_qs_w, ln_qs_b, xln);
  gemm_wmma<<<(4096 / 32) * (3072 / 64) / 8, 256, 0, stream>>>(
      xln, wqkv_p, 4096, 3072, 1024, qkvbuf, nullptr, 0);
  qkv_post<<<4096, 256, 0, stream>>>(qkvbuf, q_ln_seq_w, k_ln_seq_w, Qf, Kf, Vtf);
  zeropad_kv<<<(BATCH * NHEAD * (LPAD - LQ) * DHEAD + 255) / 256, 256, 0, stream>>>(
      Qf, Kf, Vtf);

  // struct token path (T_input == 0 -> timestep embedding is [0]*128 ++ [1]*128)
  s_hT<<<4, 256, 0, stream>>>(T_w1, T_b1, hT);
  s_htemp<<<16, 256, 0, stream>>>(temp, temp_w1, temp_b1, htmp);
  s_Te<<<4, 256, 0, stream>>>(hT, T_w2, T_b2, TeB);
  s_tempE<<<16, 256, 0, stream>>>(htmp, temp_w2, temp_b2, tmpE);
  s_xmod<<<4, 256, 0, stream>>>(x_struc, ln_qt_w, ln_qt_b, tmpE, TeB, xmod);
  s_qkvt<<<(BATCH * 3072 + 255) / 256, 256, 0, stream>>>(xmod, W_qkv_struc, qkvt);
  struct_finish<<<4, 256, 0, stream>>>(qkvt, q_ln_struc_w, k_ln_struc_w, Qf, Kf, Vtf);

  // attention (flash, WMMA)
  attn_kernel<<<BATCH * NHEAD * NQT, 32, 0, stream>>>(Qf, Kf, Vtf, ctxs, ctxt);

  // out projections + residual
  gemm_wmma<<<(4096 / 32) * (1024 / 64) / 8, 256, 0, stream>>>(
      ctxs, wout_p, 4096, 1024, 1024, out_seq, x_seq, 1);
  sdot_outstruc<<<16, 256, 0, stream>>>(ctxt, W_out_struc, x_struc, xs2);

  // seq FFN
  ln_rows_f16<<<4096, 256, 0, stream>>>(out_seq, ffn_seq_ln_w, ffn_seq_ln_b, xln);
  gemm_wmma<<<(4096 / 32) * (4096 / 64) / 8, 256, 0, stream>>>(
      xln, w1_p, 4096, 4096, 1024, hidden, nullptr, 2);
  gemm_wmma<<<(4096 / 32) * (1024 / 64) / 8, 256, 0, stream>>>(
      hidden, w2_p, 4096, 1024, 4096, out_seq, out_seq, 1);

  // struct FFN
  ln_rows_f32<<<4, 256, 0, stream>>>(xs2, ffn_st_ln_w, ffn_st_ln_b, slnb);
  sffn1<<<(BATCH * 4096 + 255) / 256, 256, 0, stream>>>(slnb, ffn_st_w1, shid);
  sffn2<<<16, 256, 0, stream>>>(shid, ffn_st_w2, xs2, out_st);
}
